// MixtralDecoderLayer_33535104647333
// MI455X (gfx1250) — compile-verified, matrix operand order checked by reading, not run on hardware
//
#include <hip/hip_runtime.h>
#include <hip/hip_bf16.h>

// ---------------- problem constants (match reference) ----------------
#define T_   2048
#define H_   2048
#define NH_  16
#define NKV_ 4
#define HD_  128
#define E_   8
#define F_   4096
#define FS_  4096

// ---------------- vector types ----------------
typedef __attribute__((ext_vector_type(16))) __bf16 v16bf;
typedef __attribute__((ext_vector_type(8)))  __bf16 v8bf;
typedef __attribute__((ext_vector_type(8)))  float  v8f;
typedef __attribute__((ext_vector_type(4)))  float  v4f;
typedef __attribute__((ext_vector_type(4)))  unsigned int v4u;
typedef __attribute__((ext_vector_type(8)))  int    v8i;
typedef __attribute__((ext_vector_type(4)))  int    v4i;

#if __has_builtin(__builtin_amdgcn_tensor_load_to_lds)
#define HAVE_TDM 1
#else
#define HAVE_TDM 0
#endif

__device__ __forceinline__ void wait_tensorcnt0() {
#if __has_builtin(__builtin_amdgcn_s_wait_tensorcnt)
  __builtin_amdgcn_s_wait_tensorcnt(0);
#else
  asm volatile("s_wait_tensorcnt 0x0" ::: "memory");
#endif
}

#if HAVE_TDM
// TDM: DMA a [tile_dim1=128 rows x tile_dim0=32 elems] fp32 tile from a
// row-major [M][K] tensor into LDS (contiguous [128][32] f32).
// D# layout per CDNA5 ISA (2-D tile: groups 2/3 zero).
__device__ __forceinline__ void tdm_load_a(const float* gptr, unsigned lds_off,
                                           int M, int K) {
  unsigned long long ga = (unsigned long long)(const void*)gptr;
  v4u g0;
  g0[0] = 1u;                       // count=1, user mode, no gather
  g0[1] = lds_off;                  // lds_addr (bytes)
  g0[2] = (unsigned)ga;             // global_addr[31:0]
  g0[3] = (unsigned)((ga >> 32) & 0x1FFFFFFu) | 0x80000000u;  // addr[56:32]|type=2
  v8i g1;
  g1[0] = 0x20000;                  // workgroup_mask=0, data_size=2 (4B)
  g1[1] = (int)(((unsigned)K & 0xFFFFu) << 16);               // tensor_dim0 lo16
  g1[2] = (int)((((unsigned)K >> 16) & 0xFFFFu) |
                (((unsigned)M & 0xFFFFu) << 16));             // dim0 hi | dim1 lo
  g1[3] = (int)((((unsigned)M >> 16) & 0xFFFFu) | (32u << 16)); // dim1 hi | tile_dim0
  g1[4] = 128;                      // tile_dim1=128, tile_dim2=0
  g1[5] = K;                        // tensor_dim0_stride[31:0] (elements)
  g1[6] = 0;                        // dim0_stride hi | dim1_stride lo
  g1[7] = 0;
  v4i z4; z4[0] = 0; z4[1] = 0; z4[2] = 0; z4[3] = 0;
#if defined(__clang_major__) && (__clang_major__ >= 23)
  v8i z8; for (int i = 0; i < 8; i++) z8[i] = 0;
  __builtin_amdgcn_tensor_load_to_lds(g0, g1, z4, z4, z8, 0);
#else
  __builtin_amdgcn_tensor_load_to_lds(g0, g1, z4, z4, 0);
#endif
}
#endif

// ---------------- WMMA GEMM: C[M,N] = epi(A[M,K] @ B[K,N]) ----------------
// Tile 128x128, K-step 32, 256 threads = 8 waves; wave -> 32x64 subtile =
// 2x4 WMMA accumulators. A tile: TDM DMA (fp32 in LDS, cvt at fragment
// load). B tile: manual load + cvt, stored transposed ([n][k]) in bf16.
#define BM 128
#define BN 128
#define BK 32

__device__ __forceinline__ v16bf load_frag_bf16(const __bf16* rowp, int lane) {
  const int kb = (lane >> 4) << 3;
  v8bf lo = *(const v8bf*)(rowp + kb);
  v8bf hi = *(const v8bf*)(rowp + kb + 16);
  return __builtin_shufflevector(lo, hi, 0,1,2,3,4,5,6,7,8,9,10,11,12,13,14,15);
}

__device__ __forceinline__ void stage_b(
    const float* __restrict__ B, int N, int n0, int k0, int tid,
    __bf16 (*dB)[BK])
{
  v4f b[4];
  int bkk[4], bnc[4];
#pragma unroll
  for (int q = 0; q < 4; q++) {
    int idx = q * 256 + tid;
    bkk[q] = idx >> 5;
    bnc[q] = (idx & 31) << 2;
    b[q] = *(const v4f*)(B + (size_t)(k0 + bkk[q]) * N + n0 + bnc[q]);
  }
#pragma unroll
  for (int q = 0; q < 4; q++) {
    dB[bnc[q] + 0][bkk[q]] = (__bf16)b[q][0];
    dB[bnc[q] + 1][bkk[q]] = (__bf16)b[q][1];
    dB[bnc[q] + 2][bkk[q]] = (__bf16)b[q][2];
    dB[bnc[q] + 3][bkk[q]] = (__bf16)b[q][3];
  }
}

#if !HAVE_TDM
__device__ __forceinline__ void stage_a_f32(
    const float* __restrict__ A, int K, int m0, int k0, int tid,
    float (*dA)[BK])
{
#pragma unroll
  for (int q = 0; q < 4; q++) {
    int idx = q * 256 + tid;
    int row = idx >> 3, kk = (idx & 7) << 2;
    *(v4f*)&dA[row][kk] = *(const v4f*)(A + (size_t)(m0 + row) * K + k0 + kk);
  }
}
#endif

__global__ __launch_bounds__(256) void gemm_bf16_wmma(
    const float* __restrict__ A, const float* __restrict__ B,
    float* __restrict__ C, int M, int N, int K,
    const float* __restrict__ add0,      // += add0[idx]
    const float* __restrict__ add1,      // += add1[idx]
    const float* __restrict__ silu_mul,  // val = acc * silu(silu_mul[idx])
    const float* __restrict__ rowscale,  // val *= rowscale[m]
    int accumulate)                      // out[idx] += val (else =)
{
  __shared__ float  sAf[2][BM][BK];   // fp32 A tiles (TDM destination)
  __shared__ __bf16 sB[2][BN][BK];    // bf16 B^T tiles

  const int tid  = threadIdx.x;
  const int lane = tid & 31;
  const int wave = tid >> 5;
  const int wm   = wave & 3;   // 4 row slots of 32
  const int wn   = wave >> 2;  // 2 col slots of 64
  const int m0   = blockIdx.y * BM;
  const int n0   = blockIdx.x * BN;

  v8f acc[2][4];
#pragma unroll
  for (int i = 0; i < 2; i++)
#pragma unroll
    for (int j = 0; j < 4; j++)
#pragma unroll
      for (int v = 0; v < 8; v++) acc[i][j][v] = 0.0f;

  const int nk = K / BK;

#if HAVE_TDM
  if (wave == 0)
    tdm_load_a(A + (size_t)m0 * K,
               (unsigned)(size_t)(void*)&sAf[0][0][0], M, K);
#else
  stage_a_f32(A, K, m0, 0, tid, sAf[0]);
#endif
  stage_b(B, N, n0, 0, tid, sB[0]);
#if HAVE_TDM
  if (wave == 0) wait_tensorcnt0();
#endif
  __syncthreads();

  for (int ks = 0; ks < nk; ks++) {
    const int buf = ks & 1;
    if (ks + 1 < nk) {
      const int k0 = (ks + 1) * BK;
#if HAVE_TDM
      if (wave == 0)
        tdm_load_a(A + (size_t)m0 * K + k0,
                   (unsigned)(size_t)(void*)&sAf[buf ^ 1][0][0], M, K);
#else
      stage_a_f32(A, K, m0, k0, tid, sAf[buf ^ 1]);
#endif
      __builtin_prefetch(B + (size_t)(k0 + BK + (tid >> 4)) * N + n0, 0, 0);
      stage_b(B, N, n0, k0, tid, sB[buf ^ 1]);
    }

    // Load RAW fp32 for BOTH A fragments first (so DS waits stagger),
    // then convert; B fragments are bf16 and feed WMMA directly.
    const int kb = (lane >> 4) << 3;
    v4f araw[2][4];
#pragma unroll
    for (int mt = 0; mt < 2; mt++) {
      const float* rowp = &sAf[buf][wm * 32 + mt * 16 + (lane & 15)][0];
      araw[mt][0] = *(const v4f*)(rowp + kb);
      araw[mt][1] = *(const v4f*)(rowp + kb + 4);
      araw[mt][2] = *(const v4f*)(rowp + kb + 16);
      araw[mt][3] = *(const v4f*)(rowp + kb + 20);
    }
    v16bf bfm[4];
#pragma unroll
    for (int nt = 0; nt < 4; nt++)
      bfm[nt] = load_frag_bf16(&sB[buf][wn * 64 + nt * 16 + (lane & 15)][0], lane);

    v16bf af[2];
#pragma unroll
    for (int mt = 0; mt < 2; mt++) {
#pragma unroll
      for (int i = 0; i < 4; i++) {
        af[mt][i]      = (__bf16)araw[mt][0][i];
        af[mt][i + 4]  = (__bf16)araw[mt][1][i];
        af[mt][i + 8]  = (__bf16)araw[mt][2][i];
        af[mt][i + 12] = (__bf16)araw[mt][3][i];
      }
    }

#pragma unroll
    for (int mt = 0; mt < 2; mt++)
#pragma unroll
      for (int nt = 0; nt < 4; nt++)
        acc[mt][nt] = __builtin_amdgcn_wmma_f32_16x16x32_bf16(
            false, af[mt], false, bfm[nt], (short)0, acc[mt][nt], false, false);

#if HAVE_TDM
    if (wave == 0) wait_tensorcnt0();   // next A tile resident before barrier
#endif
    __syncthreads();
  }

  // epilogue: C/D layout -> lane covers n=(lane&15), m = (lane>>4)*8 + vgpr
#pragma unroll
  for (int mt = 0; mt < 2; mt++) {
#pragma unroll
    for (int nt = 0; nt < 4; nt++) {
      const int mb = m0 + wm * 32 + mt * 16 + ((lane >> 4) << 3);
      const int n  = n0 + wn * 64 + nt * 16 + (lane & 15);
#pragma unroll
      for (int v = 0; v < 8; v++) {
        const int mm = mb + v;
        const size_t idx = (size_t)mm * N + n;
        float val = acc[mt][nt][v];
        if (silu_mul) {
          float g = silu_mul[idx];
          val *= g / (1.0f + __expf(-g));
        }
        if (rowscale) val *= rowscale[mm];
        float base = 0.0f;
        if (add0) base += add0[idx];
        if (add1) base += add1[idx];
        if (accumulate) base += C[idx];
        C[idx] = base + val;
      }
    }
  }
}

// ---------------- RMSNorm ----------------
__global__ __launch_bounds__(256) void rmsnorm_kernel(
    const float* __restrict__ x, const float* __restrict__ w,
    float* __restrict__ out)
{
  __shared__ float red[256];
  const int t = blockIdx.x;
  const float* xr = x + (size_t)t * H_;
  float ss = 0.0f;
  for (int i = threadIdx.x; i < H_; i += 256) { float v = xr[i]; ss += v * v; }
  red[threadIdx.x] = ss;
  __syncthreads();
  for (int st = 128; st > 0; st >>= 1) {
    if (threadIdx.x < st) red[threadIdx.x] += red[threadIdx.x + st];
    __syncthreads();
  }
  const float r = rsqrtf(red[0] / (float)H_ + 1e-6f);
  for (int i = threadIdx.x; i < H_; i += 256)
    out[(size_t)t * H_ + i] = xr[i] * r * w[i];
}

// ---------------- RoPE (in place), layout [T][heads][128] ----------------
__global__ __launch_bounds__(256) void rope_kernel(
    float* __restrict__ x, const int* __restrict__ pos, int nheads)
{
  const int i = blockIdx.x * 256 + threadIdx.x;
  const int d = i & 63;
  const int hh = (i >> 6) % nheads;
  const int t = i / (64 * nheads);
  const float inv = __expf(-13.815510558f * (float)d * (1.0f / 64.0f)); // 1e6^(-d/64)
  const float ang = (float)pos[t] * inv;
  const float c = __cosf(ang), s = __sinf(ang);
  float* p = x + (size_t)t * nheads * HD_ + hh * HD_;
  const float x1 = p[d], x2 = p[d + 64];
  p[d]      = x1 * c - x2 * s;
  p[d + 64] = x2 * c + x1 * s;
}

// ---------------- WMMA flash attention -----------------------------------
#define AW 8  // waves per block

__device__ __forceinline__ v16bf load_frag_global(const float* p, float s) {
  v4f x0 = *(const v4f*)p;
  v4f x1 = *(const v4f*)(p + 4);
  v4f x2 = *(const v4f*)(p + 16);
  v4f x3 = *(const v4f*)(p + 20);
  v16bf f;
#pragma unroll
  for (int i = 0; i < 4; i++) {
    f[i]      = (__bf16)(x0[i] * s);
    f[i + 4]  = (__bf16)(x1[i] * s);
    f[i + 8]  = (__bf16)(x2[i] * s);
    f[i + 12] = (__bf16)(x3[i] * s);
  }
  return f;
}

__device__ __forceinline__ v16bf pad_frag(v8bf lo) {
  v8bf zz;
#pragma unroll
  for (int i = 0; i < 8; i++) zz[i] = (__bf16)0.0f;
  return __builtin_shufflevector(lo, zz, 0,1,2,3,4,5,6,7,8,9,10,11,12,13,14,15);
}

__global__ __launch_bounds__(256) void attn_wmma_kernel(
    const float* __restrict__ q, const float* __restrict__ k,
    const float* __restrict__ v, float* __restrict__ out)
{
  __shared__ __bf16 sV[AW][HD_][16];   // per-wave V^T tile [hd][key]
  __shared__ __bf16 sP[AW][16][16];    // per-wave P tile [m][key]

  const int lane = threadIdx.x & 31;
  const int wv   = threadIdx.x >> 5;
  const int task = blockIdx.x * AW + wv;
  const int h    = task & (NH_ - 1);
  const int q0   = (task >> 4) << 4;
  const int kvh  = h >> 2;                 // NH/NKV = 4
  const float scale = 0.08838834764831845f; // 1/sqrt(128)

  const int lh = lane >> 4;   // lane half
  const int ln = lane & 15;
  const int kb = lh << 3;     // 0 or 8

  v16bf qf[4];
  {
    const float* qrow = q + (size_t)(q0 + ln) * (NH_ * HD_) + h * HD_;
#pragma unroll
    for (int c = 0; c < 4; c++)
      qf[c] = load_frag_global(qrow + c * 32 + kb, scale);
  }

  v8f oacc[8];
#pragma unroll
  for (int nt = 0; nt < 8; nt++)
#pragma unroll
    for (int vv = 0; vv < 8; vv++) oacc[nt][vv] = 0.0f;
  float rowM[8], rowS[8];
#pragma unroll
  for (int vv = 0; vv < 8; vv++) { rowM[vv] = -3.4e38f; rowS[vv] = 0.0f; }

  for (int s0 = 0; s0 <= q0; s0 += 16) {
    v16bf kf[4];
    const float* krow = k + (size_t)(s0 + ln) * (NKV_ * HD_) + kvh * HD_;
#pragma unroll
    for (int c = 0; c < 4; c++)
      kf[c] = load_frag_global(krow + c * 32 + kb, 1.0f);

    {
      const float* vrow = v + (size_t)(s0 + ln) * (NKV_ * HD_) + kvh * HD_ + (lh << 6);
#pragma unroll
      for (int i = 0; i < 64; i += 4) {
        v4f val = *(const v4f*)(vrow + i);
        sV[wv][(lh << 6) + i + 0][ln] = (__bf16)val[0];
        sV[wv][(lh << 6) + i + 1][ln] = (__bf16)val[1];
        sV[wv][(lh << 6) + i + 2][ln] = (__bf16)val[2];
        sV[wv][(lh << 6) + i + 3][ln] = (__bf16)val[3];
      }
    }

    v8f sacc;
#pragma unroll
    for (int vv = 0; vv < 8; vv++) sacc[vv] = 0.0f;
#pragma unroll
    for (int c = 0; c < 4; c++)
      sacc = __builtin_amdgcn_wmma_f32_16x16x32_bf16(
          false, qf[c], false, kf[c], (short)0, sacc, false, false);

    if (s0 == q0) {
#pragma unroll
      for (int vv = 0; vv < 8; vv++)
        if (ln > kb + vv) sacc[vv] = -1e30f;
    }

#pragma unroll
    for (int vv = 0; vv < 8; vv++) {
      float x = sacc[vv];
      x = fmaxf(x, __shfl_xor(x, 1, 32));
      x = fmaxf(x, __shfl_xor(x, 2, 32));
      x = fmaxf(x, __shfl_xor(x, 4, 32));
      x = fmaxf(x, __shfl_xor(x, 8, 32));
      const float nm = fmaxf(rowM[vv], x);
      const float cf = __expf(rowM[vv] - nm);
      const float pv = __expf(sacc[vv] - nm);
      float ps = pv;
      ps += __shfl_xor(ps, 1, 32);
      ps += __shfl_xor(ps, 2, 32);
      ps += __shfl_xor(ps, 4, 32);
      ps += __shfl_xor(ps, 8, 32);
      rowS[vv] = rowS[vv] * cf + ps;
      rowM[vv] = nm;
#pragma unroll
      for (int nt = 0; nt < 8; nt++) oacc[nt][vv] *= cf;
      sP[wv][kb + vv][ln] = (__bf16)pv;
    }

    __builtin_amdgcn_wave_barrier();  // keep LDS store->load ordering

    v16bf pf = pad_frag(*(const v8bf*)&sP[wv][ln][kb]);

#pragma unroll
    for (int nt = 0; nt < 8; nt++) {
      v16bf vf = pad_frag(*(const v8bf*)&sV[wv][nt * 16 + ln][kb]);
      oacc[nt] = __builtin_amdgcn_wmma_f32_16x16x32_bf16(
          false, pf, false, vf, (short)0, oacc[nt], false, false);
    }
    __builtin_amdgcn_wave_barrier();
  }

#pragma unroll
  for (int nt = 0; nt < 8; nt++)
#pragma unroll
    for (int vv = 0; vv < 8; vv++)
      out[(size_t)(q0 + kb + vv) * (NH_ * HD_) + h * HD_ + nt * 16 + ln] =
          oacc[nt][vv] / rowS[vv];
}

// ---------------- router: softmax top-2 -> combine^T [E][T], sigmoid gate
__global__ __launch_bounds__(256) void router_kernel(
    const float* __restrict__ x, const float* __restrict__ rw,
    const float* __restrict__ wsg, float* __restrict__ combT,
    float* __restrict__ sg)
{
  const int t = blockIdx.x * 256 + threadIdx.x;
  const float* xr = x + (size_t)t * H_;
  float logit[E_];
#pragma unroll
  for (int e = 0; e < E_; e++) logit[e] = 0.0f;
  float sd = 0.0f;
  for (int i = 0; i < H_; i++) {
    const float xv = xr[i];
    const float* r = rw + (size_t)i * E_;
#pragma unroll
    for (int e = 0; e < E_; e++) logit[e] += xv * r[e];
    sd += xv * wsg[i];
  }
  float mx = logit[0];
#pragma unroll
  for (int e = 1; e < E_; e++) mx = fmaxf(mx, logit[e]);
  float pr[E_], den = 0.0f;
#pragma unroll
  for (int e = 0; e < E_; e++) { pr[e] = __expf(logit[e] - mx); den += pr[e]; }
#pragma unroll
  for (int e = 0; e < E_; e++) pr[e] /= den;
  int i1 = 0;
#pragma unroll
  for (int e = 1; e < E_; e++) if (pr[e] > pr[i1]) i1 = e;
  int i2 = (i1 == 0) ? 1 : 0;
#pragma unroll
  for (int e = 0; e < E_; e++) if (e != i1 && pr[e] > pr[i2]) i2 = e;
  const float s2 = pr[i1] + pr[i2];
#pragma unroll
  for (int e = 0; e < E_; e++) combT[(size_t)e * T_ + t] = 0.0f;
  combT[(size_t)i1 * T_ + t] = pr[i1] / s2;
  combT[(size_t)i2 * T_ + t] = pr[i2] / s2;
  sg[t] = 1.0f / (1.0f + __expf(-sd));
}

// ---------------- host-side orchestration ----------------
extern "C" void kernel_launch(void* const* d_in, const int* in_sizes, int n_in,
                              void* d_out, int out_size, void* d_ws, size_t ws_size,
                              hipStream_t stream) {
  (void)in_sizes; (void)n_in; (void)out_size; (void)ws_size;
  const float* hidden = (const float*)d_in[0];
  const int*   pos    = (const int*)d_in[1];
  const float* ln1    = (const float*)d_in[2];
  const float* ln2    = (const float*)d_in[3];
  const float* wq     = (const float*)d_in[4];
  const float* wk     = (const float*)d_in[5];
  const float* wv     = (const float*)d_in[6];
  const float* wo     = (const float*)d_in[7];
  const float* rw     = (const float*)d_in[8];
  const float* wg     = (const float*)d_in[9];
  const float* wu     = (const float*)d_in[10];
  const float* wd     = (const float*)d_in[11];
  const float* wsgate = (const float*)d_in[12];
  const float* wsup   = (const float*)d_in[13];
  const float* wsdown = (const float*)d_in[14];
  const float* wsig   = (const float*)d_in[15];
  float* out = (float*)d_out;
  float* ws  = (float*)d_ws;

  size_t o = 0;
  float* xn    = ws + o; o += (size_t)T_ * H_;
  float* qb    = ws + o; o += (size_t)T_ * NH_ * HD_;
  float* kbuf  = ws + o; o += (size_t)T_ * NKV_ * HD_;
  float* vbuf  = ws + o; o += (size_t)T_ * NKV_ * HD_;
  float* attn  = ws + o; o += (size_t)T_ * NH_ * HD_;
  float* hid2  = ws + o; o += (size_t)T_ * H_;
  float* combT = ws + o; o += (size_t)E_ * T_;
  float* sgb   = ws + o; o += (size_t)T_;
  float* gbuf  = ws + o; o += (size_t)T_ * F_;
  float* gub   = ws + o; o += (size_t)T_ * F_;
  float* moe   = ws + o; o += (size_t)T_ * H_;

  dim3 blk(256);
  auto gemm = [&](const float* A, const float* B, float* C, int M, int N, int K,
                  const float* a0, const float* a1, const float* sm,
                  const float* rs, int accu) {
    dim3 g(N / BN, M / BM);
    gemm_bf16_wmma<<<g, blk, 0, stream>>>(A, B, C, M, N, K, a0, a1, sm, rs, accu);
  };

  // attn block
  rmsnorm_kernel<<<T_, blk, 0, stream>>>(hidden, ln1, xn);
  gemm(xn, wq, qb,   T_, NH_ * HD_,  H_, nullptr, nullptr, nullptr, nullptr, 0);
  gemm(xn, wk, kbuf, T_, NKV_ * HD_, H_, nullptr, nullptr, nullptr, nullptr, 0);
  gemm(xn, wv, vbuf, T_, NKV_ * HD_, H_, nullptr, nullptr, nullptr, nullptr, 0);
  rope_kernel<<<(T_ * NH_ * 64) / 256, blk, 0, stream>>>(qb, pos, NH_);
  rope_kernel<<<(T_ * NKV_ * 64) / 256, blk, 0, stream>>>(kbuf, pos, NKV_);
  attn_wmma_kernel<<<(T_ / 16) * NH_ / AW, blk, 0, stream>>>(qb, kbuf, vbuf, attn);
  gemm(attn, wo, hid2, T_, H_, NH_ * HD_, hidden, nullptr, nullptr, nullptr, 0);

  // MoE + shared FFN block
  rmsnorm_kernel<<<T_, blk, 0, stream>>>(hid2, ln2, xn);
  router_kernel<<<T_ / 256, blk, 0, stream>>>(xn, rw, wsig, combT, sgb);
  for (int e = 0; e < E_; e++) {
    const float* wge = wg + (size_t)e * H_ * F_;
    const float* wue = wu + (size_t)e * H_ * F_;
    const float* wde = wd + (size_t)e * F_ * H_;
    gemm(xn, wge, gbuf, T_, F_, H_, nullptr, nullptr, nullptr, nullptr, 0);
    gemm(xn, wue, gub,  T_, F_, H_, nullptr, nullptr, gbuf, nullptr, 0);       // silu(g)*u
    gemm(gub, wde, moe, T_, H_, F_, nullptr, nullptr, nullptr,
         combT + (size_t)e * T_, e > 0);                                       // += combine*down
  }
  gemm(xn, wsgate, gbuf, T_, FS_, H_, nullptr, nullptr, nullptr, nullptr, 0);
  gemm(xn, wsup,   gub,  T_, FS_, H_, nullptr, nullptr, gbuf, nullptr, 0);
  gemm(gub, wsdown, out, T_, H_, FS_, hid2, moe, nullptr, sgb, 0);             // res+moe+sg*so
}
